// AutoregressiveModel_47536698032720
// MI455X (gfx1250) — matrix operand order repaired and checked by reading, not compiled, not run on hardware
//
#include <hip/hip_runtime.h>
#include <cstdint>
#include <cstddef>

// ---------------------------------------------------------------------------
// Problem constants
// ---------------------------------------------------------------------------
#define B_   64
#define T_   512
#define H_   512
#define G4H  2048          // 4*H gate width
#define MT   32768         // B*T

#define AS1 __attribute__((address_space(1)))
#define AS3 __attribute__((address_space(3)))

typedef _Float16 h16;
typedef _Float16 v16h __attribute__((ext_vector_type(16)));
typedef _Float16 v8h  __attribute__((ext_vector_type(8)));
typedef float    v8f  __attribute__((ext_vector_type(8)));
typedef unsigned int u32x4 __attribute__((ext_vector_type(4)));
typedef int          i32x4 __attribute__((ext_vector_type(4)));
typedef int          i32x8 __attribute__((ext_vector_type(8)));

// D = A(16x32 f16) * B(32x16 f16) + C(16x16 f32)
__device__ __forceinline__ v8f wmma16(v16h a, v16h b, v8f c) {
  return __builtin_amdgcn_wmma_f32_16x16x32_f16(
      /*neg_a=*/false, a, /*neg_b=*/false, b,
      /*c_mod=*/(short)0, c, /*reuse_a=*/false, /*reuse_b=*/false);
}

// A-fragment (16x32, 16-bit): lane holds K[2kb..2kb+15] (contiguous)
// with kb = (lane>=16)?8:0.  'row' points at A[m][kk] (row-major, any stride).
__device__ __forceinline__ v16h ldA(const h16* row, int kb) {
  v16h a;
  ((v8h*)&a)[0] = *(const v8h*)(row + 2 * kb);
  ((v8h*)&a)[1] = *(const v8h*)(row + 2 * kb + 8);
  return a;
}

__device__ __forceinline__ float sigf(float x)  { return 1.f / (1.f + __expf(-x)); }
__device__ __forceinline__ float tanhf_(float x){ return 2.f / (1.f + __expf(-2.f * x)) - 1.f; }
__device__ __forceinline__ float seluf(float x) {
  const float a = 1.6732632423543772f, s = 1.0507009873554805f;
  return x > 0.f ? s * x : s * a * (__expf(x) - 1.f);
}

// ---------------------------------------------------------------------------
// Tensor Data Mover: load a [rows x cols] f16 tile (row stride in elements)
// from global to LDS (packed row-major).  D# layout per CDNA5 ISA ch.8.
// Must be executed by exactly one wave (TDM ignores EXEC).
// ---------------------------------------------------------------------------
#if __has_builtin(__builtin_amdgcn_tensor_load_to_lds)
#define HAVE_TDM 1
__device__ __forceinline__ void tdm_load_tile_f16(
    uint32_t lds_byte_addr, const void* gaddr,
    uint32_t rows, uint32_t cols, uint32_t row_stride_elems) {
  uint64_t ga = (uint64_t)(uintptr_t)gaddr;
  u32x4 g0;
  g0[0] = 1u;                                          // count=1 (valid), user mode
  g0[1] = lds_byte_addr;                               // lds_addr
  g0[2] = (uint32_t)(ga & 0xffffffffu);                // global_addr[31:0]
  g0[3] = (uint32_t)((ga >> 32) & 0x01ffffffu)         // global_addr[56:32]
        | (2u << 30);                                  // type=2 ("image")
  i32x8 g1;
  g1[0] = (int)(1u << 16);                             // data_size=1 (2B), wg_mask=0
  g1[1] = (int)(cols << 16);                           // tensor_dim0[15:0]
  g1[2] = (int)((cols >> 16) | (rows << 16));          // dim0[31:16] | dim1[15:0]
  g1[3] = (int)((rows >> 16) | (cols << 16));          // dim1[31:16] | tile_dim0
  g1[4] = (int)rows;                                   // tile_dim1 | tile_dim2=0
  g1[5] = (int)row_stride_elems;                       // tensor_dim0_stride[31:0]
  g1[6] = 0;                                           // stride[47:32] | dim1_stride lo
  g1[7] = 0;
  i32x4 z4 = {0, 0, 0, 0};
#if __clang_major__ >= 23
  i32x8 z8 = {0, 0, 0, 0, 0, 0, 0, 0};
  __builtin_amdgcn_tensor_load_to_lds(g0, g1, z4, z4, z8, 0);
#else
  __builtin_amdgcn_tensor_load_to_lds(g0, g1, z4, z4, 0);
#endif
}
#else
#define HAVE_TDM 0
#endif

// ---------------------------------------------------------------------------
// Small prep kernels
// ---------------------------------------------------------------------------
__global__ void cvt_pad_kernel(const float* __restrict__ src, h16* __restrict__ dst,
                               int rows, int K, int Kpad) {
  int idx = blockIdx.x * 256 + threadIdx.x;
  if (idx >= rows * Kpad) return;
  int r = idx / Kpad, c = idx - r * Kpad;
  dst[idx] = (c < K) ? (h16)src[(size_t)r * K + c] : (h16)0.f;
}

__global__ void bias_sum_kernel(const float* __restrict__ a, const float* __restrict__ b,
                                float* __restrict__ o, int n) {
  int i = blockIdx.x * 256 + threadIdx.x;
  if (i < n) o[i] = a[i] + b[i];
}

// P1 [MT x 96] = [static(64) | recurrent(16) | zpad(16)]  (f32 -> f16)
__global__ void pack1_kernel(const float* __restrict__ st, const float* __restrict__ rec,
                             h16* __restrict__ P) {
  int idx = blockIdx.x * 256 + threadIdx.x;
  if (idx >= MT * 96) return;
  int row = idx / 96, c = idx - row * 96;
  int b = row >> 9;                       // row / T_
  float v = 0.f;
  if (c < 64)      v = st[b * 64 + c];
  else if (c < 80) v = rec[(size_t)row * 16 + (c - 64)];
  P[idx] = (h16)v;
}

// P2 [MT x 608] = [h1(512) | static(64) | recurrent(16) | zpad(16)]
__global__ void pack2_kernel(const h16* __restrict__ h1, const float* __restrict__ st,
                             const float* __restrict__ rec, h16* __restrict__ P) {
  int idx = blockIdx.x * 256 + threadIdx.x;
  if (idx >= MT * 608) return;
  int row = idx / 608, c = idx - row * 608;
  int b = row >> 9;
  if (c < 512) { P[idx] = h1[(size_t)row * 512 + c]; return; }
  float v = 0.f;
  if (c < 576)      v = st[b * 64 + (c - 512)];
  else if (c < 592) v = rec[(size_t)row * 16 + (c - 576)];
  P[idx] = (h16)v;
}

// P [MT x 1024] = [X(512) | Y(512)]  (vectorized 16B copies)
__global__ void pack_pair_kernel(const h16* __restrict__ X, const h16* __restrict__ Y,
                                 h16* __restrict__ P) {
  int idx = blockIdx.x * 256 + threadIdx.x;   // MT*128 chunks of 8 halves
  if (idx >= MT * 128) return;
  int row = idx >> 7, q = idx & 127;
  const h16* s = (q < 64) ? &X[(size_t)row * 512 + q * 8]
                          : &Y[(size_t)row * 512 + (q - 64) * 8];
  *(uint4*)&P[(size_t)row * 1024 + q * 8] = *(const uint4*)s;
}

// ---------------------------------------------------------------------------
// Init MLP: static[64] -> 32 -> 32 -> 512 (SELU), h-path and c-path
// ---------------------------------------------------------------------------
__global__ __launch_bounds__(256) void mlp_init_kernel(
    const float* __restrict__ st,
    const float* W1h, const float* b1h, const float* W2h, const float* b2h,
    const float* W3h, const float* b3h,
    const float* W1c, const float* b1c, const float* W2c, const float* b2c,
    const float* W3c, const float* b3c,
    h16* __restrict__ h0, float* __restrict__ c0) {
  __shared__ float si[64], t1[32], t2[32];
  int b = blockIdx.x, tid = threadIdx.x;
  if (tid < 64) si[tid] = st[b * 64 + tid];
  __syncthreads();
  for (int pass = 0; pass < 2; ++pass) {
    const float *W1 = pass ? W1c : W1h, *B1 = pass ? b1c : b1h;
    const float *W2 = pass ? W2c : W2h, *B2 = pass ? b2c : b2h;
    const float *W3 = pass ? W3c : W3h, *B3 = pass ? b3c : b3h;
    if (tid < 32) {
      float s = B1[tid];
      for (int k = 0; k < 64; ++k) s += W1[tid * 64 + k] * si[k];
      t1[tid] = seluf(s);
    }
    __syncthreads();
    if (tid < 32) {
      float s = B2[tid];
      for (int k = 0; k < 32; ++k) s += W2[tid * 32 + k] * t1[k];
      t2[tid] = seluf(s);
    }
    __syncthreads();
    for (int j = tid; j < 512; j += 256) {
      float s = B3[j];
      for (int k = 0; k < 32; ++k) s += W3[j * 32 + k] * t2[k];
      float v = seluf(s);
      if (pass == 0) h0[b * 512 + j] = (h16)v; else c0[b * 512 + j] = v;
    }
    __syncthreads();
  }
}

// ---------------------------------------------------------------------------
// Time-parallel input projection: C[MT x 2048] = A[MT x K] * W[2048 x K]^T + bias
// 128x128 block tile, 8 waves (2x4), each wave 64x32 via 4x2 WMMA accumulators.
// ---------------------------------------------------------------------------
__global__ __launch_bounds__(256) void gemm_xproj_kernel(
    const h16* __restrict__ A, int lda,
    const h16* __restrict__ W, int ldw,
    const float* __restrict__ bias,
    float* __restrict__ C, int ldc, int K) {
  __shared__ h16 la[128 * 32];
  __shared__ h16 lb[128 * 32];
  const int tid = threadIdx.x;
  const int m0 = blockIdx.y * 128;
  const int n0 = blockIdx.x * 128;
  const int wave = tid >> 5, lane = tid & 31;
  const int wm = wave & 1, wn = wave >> 1;          // 2 x 4 wave grid
  const int am = lane & 15, akb = (lane >> 4) << 3; // A-frag lane params
  const int bn = lane & 15, bkb = (lane >> 4) << 4; // B-frag lane params

  // Hoisted staging addresses: each thread owns 2 rows (r, r+64) at column sc.
  const int sr = tid >> 2, sc = (tid & 3) << 3;
  const h16* pa0 = A + (size_t)(m0 + sr) * lda + sc;
  const h16* pa1 = A + (size_t)(m0 + 64 + sr) * lda + sc;
  const h16* pw0 = W + (size_t)(n0 + sr) * ldw + sc;
  const h16* pw1 = W + (size_t)(n0 + 64 + sr) * ldw + sc;
  h16* qa0 = &la[sr * 32 + sc];
  h16* qa1 = &la[(64 + sr) * 32 + sc];
  h16* qb0 = &lb[sr * 32 + sc];
  h16* qb1 = &lb[(64 + sr) * 32 + sc];

  v8f acc[4][2] = {};
  for (int kk = 0; kk < K; kk += 32) {
    __syncthreads();
    *(uint4*)qa0 = *(const uint4*)(pa0 + kk);
    *(uint4*)qa1 = *(const uint4*)(pa1 + kk);
    *(uint4*)qb0 = *(const uint4*)(pw0 + kk);
    *(uint4*)qb1 = *(const uint4*)(pw1 + kk);
    __syncthreads();
    v16h bf[2];
#pragma unroll
    for (int j = 0; j < 2; ++j)
      bf[j] = *(const v16h*)&lb[(wn * 32 + j * 16 + bn) * 32 + bkb];
#pragma unroll
    for (int i = 0; i < 4; ++i) {
      v16h af = ldA(&la[(wm * 64 + i * 16 + am) * 32], akb);
#pragma unroll
      for (int j = 0; j < 2; ++j) acc[i][j] = wmma16(af, bf[j], acc[i][j]);
    }
  }
#pragma unroll
  for (int i = 0; i < 4; ++i)
#pragma unroll
    for (int j = 0; j < 2; ++j) {
      int n = n0 + wn * 32 + j * 16 + (lane & 15);
      float bv = bias[n];
#pragma unroll
      for (int r = 0; r < 8; ++r) {
        int row = m0 + wm * 64 + i * 16 + r + ((lane >> 4) << 3);
        C[(size_t)row * ldc + n] = acc[i][j][r] + bv;
      }
    }
}

// ---------------------------------------------------------------------------
// Persistent LSTM scan. 64 WGs = 4 batch groups (16 rows) x 16 gate slices
// (32 hidden units -> 128 gate cols).  Whh slice (128x512 f16 = 128KB) loaded
// once per layer via the Tensor Data Mover and reused for all 512 steps;
// h ping-pongs through L2 with one 16-WG barrier per step; c in registers.
// ---------------------------------------------------------------------------
#define SCAN_LW_BYTES (128 * 512 * 2)   // 131072
#define SCAN_LH_BYTES (16 * 512 * 2)    // 16384
#define SCAN_LG_BYTES (16 * 128 * 4)    // 8192
#define SCAN_SHMEM    (SCAN_LW_BYTES + SCAN_LH_BYTES + SCAN_LG_BYTES)

__global__ __launch_bounds__(256, 1) void lstm_scan_kernel(
    const h16* __restrict__ Whh,     // [2048 x 512] f16, gate-row major
    const float* __restrict__ xproj, // [MT x 2048] f32 (includes bih+bhh)
    const h16* __restrict__ h0,      // [64 x 512] f16
    const float* __restrict__ c0,    // [64 x 512] f32
    h16* __restrict__ hb0, h16* __restrict__ hb1,   // ping-pong h [64 x 512]
    h16* __restrict__ hhist,         // [MT x 512] f16
    float* __restrict__ hN, float* __restrict__ cN, // [64 x 512] f32
    int* __restrict__ bar) {
  extern __shared__ char smem[];
  h16*   lw = (h16*)smem;
  h16*   lh = (h16*)(smem + SCAN_LW_BYTES);
  float* lg = (float*)(smem + SCAN_LW_BYTES + SCAN_LH_BYTES);

  const int tid = threadIdx.x;
  const int g   = blockIdx.x >> 4;   // batch group 0..3
  const int p   = blockIdx.x & 15;   // hidden slice 0..15
  const int m0  = g * 16;
  const int u0  = p * 32;

  // --- Stage Whh slice: lw row r <-> gate col (r/32)*512 + u0 + (r%32) ---
#if HAVE_TDM
  if (tid < 32) {  // one wave: TDM ignores EXEC, reach it with a scalar branch
    uint32_t lwoff = (uint32_t)(uintptr_t)(AS3 void*)lw;
#pragma unroll
    for (int cc = 0; cc < 4; ++cc) {
      const h16* gsrc = Whh + (size_t)(cc * 512 + u0) * 512;
      tdm_load_tile_f16(lwoff + (uint32_t)cc * 32 * 512 * 2, gsrc,
                        /*rows=*/32, /*cols=*/512, /*row_stride=*/512);
    }
    __builtin_amdgcn_s_wait_tensorcnt(0);
  }
#else
  for (int i = tid; i < 128 * 64; i += 256) {
    int r = i >> 6, q = i & 63;
    int gr = (r >> 5) * 512 + u0 + (r & 31);
    *(uint4*)&lw[r * 512 + q * 8] = *(const uint4*)&Whh[(size_t)gr * 512 + q * 8];
  }
#endif

  // Cell state in registers: items tid and tid+256 of 512 (row, unit) pairs
  float creg0 = c0[(m0 + (tid >> 5)) * 512 + u0 + (tid & 31)];
  float creg1 = c0[(m0 + ((tid + 256) >> 5)) * 512 + u0 + (tid & 31)];

  const int wave = tid >> 5, lane = tid & 31;
  const int am  = lane & 15, akb = (lane >> 4) << 3;
  const int bc  = wave * 16 + (lane & 15);     // tile col (= lw row)
  const int bkb = (lane >> 4) << 4;

  for (int t = 0; t < T_; ++t) {
    const h16* hprev = (t == 0) ? h0 : ((t & 1) ? hb0 : hb1);
    __syncthreads();
    // Stage h_prev rows m0..m0+15 (16KB) into LDS
#if __has_builtin(__builtin_amdgcn_global_load_async_to_lds_b128)
    for (int i = tid; i < 1024; i += 256) {
      int r = i >> 6, q = i & 63;
      __builtin_amdgcn_global_load_async_to_lds_b128(
          (AS1 i32x4*)&hprev[(size_t)(m0 + r) * 512 + q * 8],
          (AS3 i32x4*)&lh[r * 512 + q * 8], 0, 0);
    }
#if __has_builtin(__builtin_amdgcn_s_wait_asynccnt)
    __builtin_amdgcn_s_wait_asynccnt(0);
#else
    asm volatile("s_wait_asynccnt 0x0" ::: "memory");
#endif
#else
    for (int i = tid; i < 1024; i += 256) {
      int r = i >> 6, q = i & 63;
      *(uint4*)&lh[r * 512 + q * 8] = *(const uint4*)&hprev[(size_t)(m0 + r) * 512 + q * 8];
    }
#endif
    __syncthreads();

    // gates[16 x 128] = h_prev[16 x 512] @ Whh_slice^T : 16 WMMAs per wave
    v8f acc = {};
    const h16* arow = lh + (size_t)am * 512;
    const h16* brow = lw + (size_t)bc * 512;
#pragma unroll
    for (int kk = 0; kk < 512; kk += 32) {
      v16h a = ldA(arow + kk, akb);
      v16h b = *(const v16h*)(brow + kk + bkb);
      acc = wmma16(a, b, acc);
    }
#pragma unroll
    for (int i = 0; i < 8; ++i)
      lg[(i + ((lane >> 4) << 3)) * 128 + bc] = acc[i];
    __syncthreads();

    // Elementwise gate math, c/h update
    h16* hb = (t & 1) ? hb1 : hb0;
#pragma unroll
    for (int q = 0; q < 2; ++q) {
      int it  = tid + q * 256;
      int row = it >> 5, ul = it & 31;
      const float* xp = xproj + ((size_t)((m0 + row) * T_ + t)) * G4H + u0 + ul;
      float gi = lg[row * 128 +      ul] + xp[0];
      float gf = lg[row * 128 + 32 + ul] + xp[512];
      float gg = lg[row * 128 + 64 + ul] + xp[1024];
      float go = lg[row * 128 + 96 + ul] + xp[1536];
      float c  = q ? creg1 : creg0;
      c = sigf(gf) * c + sigf(gi) * tanhf_(gg);
      float h = sigf(go) * tanhf_(c);
      if (q) creg1 = c; else creg0 = c;
      h16 hv = (h16)h;
      hb[(size_t)(m0 + row) * 512 + u0 + ul] = hv;
      hhist[((size_t)((m0 + row) * T_ + t)) * 512 + u0 + ul] = hv;
      if (t + 8 < T_) {                      // -> global_prefetch_b8
        __builtin_prefetch(xp + 8 * G4H, 0, 0);
        __builtin_prefetch(xp + 8 * G4H + 1024, 0, 0);
      }
      if (t == T_ - 1) {
        hN[(size_t)(m0 + row) * 512 + u0 + ul] = h;
        cN[(size_t)(m0 + row) * 512 + u0 + ul] = c;
      }
    }

    // 16-WG group barrier (release/acquire through L2)
    __threadfence();
    __syncthreads();
    if (tid == 0) {
      __hip_atomic_fetch_add(&bar[g], 1, __ATOMIC_RELEASE, __HIP_MEMORY_SCOPE_AGENT);
      int target = 16 * (t + 1);
      while (__hip_atomic_load(&bar[g], __ATOMIC_ACQUIRE, __HIP_MEMORY_SCOPE_AGENT) < target)
        __builtin_amdgcn_s_sleep(1);
    }
    __syncthreads();
    __threadfence();
  }
}

// ---------------------------------------------------------------------------
// Output head: out[MT x 16] = [h3|h4][MT x 1024] @ Wout^T + bout, sigmoid(:2)
// ---------------------------------------------------------------------------
__global__ __launch_bounds__(256) void gemm_out_kernel(
    const h16* __restrict__ A, const h16* __restrict__ Wo,
    const float* __restrict__ bo, float* __restrict__ out) {
  const int tid = threadIdx.x;
  const int wave = tid >> 5, lane = tid & 31;
  const int m0 = blockIdx.x * 128 + wave * 16;
  const int am = lane & 15, akb = (lane >> 4) << 3;
  const int bn = lane & 15, bkb = (lane >> 4) << 4;
  v8f acc = {};
  const h16* arow = A + (size_t)(m0 + am) * 1024;
  const h16* brow = Wo + (size_t)bn * 1024;
  for (int kk = 0; kk < 1024; kk += 32) {
    v16h a = ldA(arow + kk, akb);
    v16h b = *(const v16h*)(brow + kk + bkb);
    acc = wmma16(a, b, acc);
  }
  int n = lane & 15;
  float bv = bo[n];
#pragma unroll
  for (int r = 0; r < 8; ++r) {
    int row = m0 + r + ((lane >> 4) << 3);
    float v = acc[r] + bv;
    if (n < 2) v = 1.f / (1.f + __expf(-v));
    out[(size_t)row * 16 + n] = v;
  }
}

// ---------------------------------------------------------------------------
// Host orchestration
// ---------------------------------------------------------------------------
extern "C" void kernel_launch(void* const* d_in, const int* in_sizes, int n_in,
                              void* d_out, int out_size, void* d_ws, size_t ws_size,
                              hipStream_t stream) {
  (void)in_sizes; (void)n_in; (void)out_size; (void)ws_size;
  const float* st  = (const float*)d_in[0];
  const float* rec = (const float*)d_in[1];
  const float *Wh1 = (const float*)d_in[2],  *bh1 = (const float*)d_in[3];
  const float *Wh2 = (const float*)d_in[4],  *bh2 = (const float*)d_in[5];
  const float *Wh3 = (const float*)d_in[6],  *bh3 = (const float*)d_in[7];
  const float *Wc1 = (const float*)d_in[8],  *bc1 = (const float*)d_in[9];
  const float *Wc2 = (const float*)d_in[10], *bc2 = (const float*)d_in[11];
  const float *Wc3 = (const float*)d_in[12], *bc3 = (const float*)d_in[13];
  const float* Wih[4] = {(const float*)d_in[14], (const float*)d_in[18],
                         (const float*)d_in[22], (const float*)d_in[26]};
  const float* Whh[4] = {(const float*)d_in[15], (const float*)d_in[19],
                         (const float*)d_in[23], (const float*)d_in[27]};
  const float* bih[4] = {(const float*)d_in[16], (const float*)d_in[20],
                         (const float*)d_in[24], (const float*)d_in[28]};
  const float* bhh[4] = {(const float*)d_in[17], (const float*)d_in[21],
                         (const float*)d_in[25], (const float*)d_in[29]};
  const float* Wout = (const float*)d_in[30];
  const float* bout = (const float*)d_in[31];

  char* ws = (char*)d_ws;
  size_t off = 0;
  auto alloc = [&](size_t bytes) -> void* {
    void* pp = ws + off;
    off = (off + bytes + 255) & ~(size_t)255;
    return pp;
  };

  const int Kp[4]  = {96, 608, 1024, 1024};
  const int Kin[4] = {80, 592, 1024, 1024};
  h16* wihf[4]; for (int l = 0; l < 4; ++l) wihf[l] = (h16*)alloc((size_t)G4H * Kp[l] * 2);
  h16* whhf[4]; for (int l = 0; l < 4; ++l) whhf[l] = (h16*)alloc((size_t)G4H * 512 * 2);
  h16* woutf = (h16*)alloc((size_t)16 * 1024 * 2);
  float* bsum[4]; for (int l = 0; l < 4; ++l) bsum[l] = (float*)alloc((size_t)G4H * 4);
  h16*   h0f = (h16*)alloc((size_t)B_ * H_ * 2);
  float* c0f = (float*)alloc((size_t)B_ * H_ * 4);
  h16*   hb0 = (h16*)alloc((size_t)B_ * H_ * 2);
  h16*   hb1 = (h16*)alloc((size_t)B_ * H_ * 2);
  h16* hh[4]; for (int l = 0; l < 4; ++l) hh[l] = (h16*)alloc((size_t)MT * 512 * 2);
  h16*   P  = (h16*)alloc((size_t)MT * 1024 * 2);
  float* XP = (float*)alloc((size_t)MT * (size_t)G4H * 4);
  int*   bar = (int*)alloc(256);

  // Weight prep (f32 -> f16, K zero-padded; combined gate biases)
  for (int l = 0; l < 4; ++l) {
    int tot = G4H * Kp[l];
    cvt_pad_kernel<<<(tot + 255) / 256, 256, 0, stream>>>(Wih[l], wihf[l], G4H, Kin[l], Kp[l]);
    tot = G4H * 512;
    cvt_pad_kernel<<<(tot + 255) / 256, 256, 0, stream>>>(Whh[l], whhf[l], G4H, 512, 512);
    bias_sum_kernel<<<8, 256, 0, stream>>>(bih[l], bhh[l], bsum[l], G4H);
  }
  cvt_pad_kernel<<<(16 * 1024 + 255) / 256, 256, 0, stream>>>(Wout, woutf, 16, 1024, 1024);

  // Shared (h0, c0)
  mlp_init_kernel<<<64, 256, 0, stream>>>(st, Wh1, bh1, Wh2, bh2, Wh3, bh3,
                                          Wc1, bc1, Wc2, bc2, Wc3, bc3, h0f, c0f);

  (void)hipFuncSetAttribute(reinterpret_cast<const void*>(&lstm_scan_kernel),
                            hipFuncAttributeMaxDynamicSharedMemorySize, SCAN_SHMEM);

  float* outp = (float*)d_out;
  float *hN[4], *cN[4];
  for (int l = 0; l < 4; ++l) {
    hN[l] = outp + 524288 + (size_t)(2 * l) * 32768;
    cN[l] = hN[l] + 32768;
  }

  dim3 ggrid(G4H / 128, MT / 128);

  // Layer 1
  pack1_kernel<<<(MT * 96 + 255) / 256, 256, 0, stream>>>(st, rec, P);
  gemm_xproj_kernel<<<ggrid, 256, 0, stream>>>(P, 96, wihf[0], 96, bsum[0], XP, G4H, 96);
  (void)hipMemsetAsync(bar, 0, 256, stream);
  lstm_scan_kernel<<<64, 256, SCAN_SHMEM, stream>>>(whhf[0], XP, h0f, c0f, hb0, hb1,
                                                    hh[0], hN[0], cN[0], bar);
  // Layer 2: input [h1 | x]
  pack2_kernel<<<(MT * 608 + 255) / 256, 256, 0, stream>>>(hh[0], st, rec, P);
  gemm_xproj_kernel<<<ggrid, 256, 0, stream>>>(P, 608, wihf[1], 608, bsum[1], XP, G4H, 608);
  (void)hipMemsetAsync(bar, 0, 256, stream);
  lstm_scan_kernel<<<64, 256, SCAN_SHMEM, stream>>>(whhf[1], XP, h0f, c0f, hb0, hb1,
                                                    hh[1], hN[1], cN[1], bar);
  // Layer 3: input [h1 | h2]
  pack_pair_kernel<<<(MT * 128 + 255) / 256, 256, 0, stream>>>(hh[0], hh[1], P);
  gemm_xproj_kernel<<<ggrid, 256, 0, stream>>>(P, 1024, wihf[2], 1024, bsum[2], XP, G4H, 1024);
  (void)hipMemsetAsync(bar, 0, 256, stream);
  lstm_scan_kernel<<<64, 256, SCAN_SHMEM, stream>>>(whhf[2], XP, h0f, c0f, hb0, hb1,
                                                    hh[2], hN[2], cN[2], bar);
  // Layer 4: input [h2 | h3]
  pack_pair_kernel<<<(MT * 128 + 255) / 256, 256, 0, stream>>>(hh[1], hh[2], P);
  gemm_xproj_kernel<<<ggrid, 256, 0, stream>>>(P, 1024, wihf[3], 1024, bsum[3], XP, G4H, 1024);
  (void)hipMemsetAsync(bar, 0, 256, stream);
  lstm_scan_kernel<<<64, 256, SCAN_SHMEM, stream>>>(whhf[3], XP, h0f, c0f, hb0, hb1,
                                                    hh[3], hN[3], cN[3], bar);
  // Output head on [h3 | h4]
  pack_pair_kernel<<<(MT * 128 + 255) / 256, 256, 0, stream>>>(hh[2], hh[3], P);
  gemm_out_kernel<<<MT / 128, 256, 0, stream>>>(P, woutf, bout, outp);
}
